// RBFNet_45226005627083
// MI455X (gfx1250) — matrix operand order repaired and checked by reading, not compiled, not run on hardware
//
#include <hip/hip_runtime.h>

// ============================================================================
// RBF network, fused, for MI455X (gfx1250, wave32, WMMA).
//
// Roofline: GEMM1 34.4 GF + GEMM2 17.2 GF, HBM floor ~97MB (~4.2us @23.3TB/s)
// -> compute bound; phi (256MB) must stay on-chip -> single fused kernel.
// Precision: f16 hi/lo split (x*c ~= xh*ch + xh*cl + xl*ch) gives ~2^-21
// relative accuracy on the cross term at 3x f16-WMMA cost, ~2.7x faster than
// exact f32 WMMA (16x16x4). GEMM2 in plain f16 (phi in [0,1], |W|~0.03).
//
// Tiling: 512 WGs x 256 thr (8 waves). Wave owns 16 N-rows x 128 out-cols
// (8 f32 C-tiles). x resides in registers as f16 hi/lo A-fragments (read
// once, reused across all 1024 centers). Centers stream through LDS in
// B-fragment-swizzled layout; W chunk staged with stride-40 f16 padding for
// conflict-free fragment reads; phi transposed C->A layout via a per-wave
// LDS bounce (same-wave LDS is in-order -> no barrier).
// ============================================================================

typedef __attribute__((ext_vector_type(16))) _Float16 v16h;
typedef __attribute__((ext_vector_type(8)))  float    v8f;

#define N_ROWS   65536
#define INDIM    256
#define NCENT    1024
#define OUTDIM   128

#define WG_THREADS  256
#define WAVES       8
#define ROWS_PER_WG 128          // 16 rows per wave
#define MT          32           // centers per chunk
#define NCHUNK      (NCENT / MT) // 32

union V16U { uint4 u[2]; v16h v; };
union H8U  { _Float16 h[8]; uint4 u; };

// ---------------------------------------------------------------------------
// Kernel 1: row squared-norms of x (65536 rows) and centers (1024 rows).
// One wave32 per row: 8 floats/lane (2x float4), shuffle-reduce.
// ---------------------------------------------------------------------------
__global__ __launch_bounds__(256) void rbf_sq_kernel(
    const float* __restrict__ x, const float* __restrict__ c,
    float* __restrict__ xsq, float* __restrict__ csq) {
  const int lane = threadIdx.x & 31;
  const int wv   = threadIdx.x >> 5;
  int row = blockIdx.x * 8 + wv;               // 0 .. 66559
  const float* src;
  float* dst;
  if (row < N_ROWS) { src = x + (size_t)row * INDIM; dst = xsq + row; }
  else { int r = row - N_ROWS; src = c + (size_t)r * INDIM; dst = csq + r; }
  float4 p = *(const float4*)(src + lane * 8);
  float4 q = *(const float4*)(src + lane * 8 + 4);
  float s = p.x*p.x + p.y*p.y + p.z*p.z + p.w*p.w
          + q.x*q.x + q.y*q.y + q.z*q.z + q.w*q.w;
  #pragma unroll
  for (int off = 16; off > 0; off >>= 1) s += __shfl_down(s, off, 32);
  if (lane == 0) *dst = s;
}

// ---------------------------------------------------------------------------
// Kernel 2: fused  out = exp(-sigma*(xsq+csq-2*x@c^T)) @ W^T + b
// ---------------------------------------------------------------------------
__global__ __launch_bounds__(256) void rbf_fused_kernel(
    const float* __restrict__ x, const float* __restrict__ centers,
    const float* __restrict__ sigmas, const float* __restrict__ W,
    const float* __restrict__ bias, const float* __restrict__ xsq,
    const float* __restrict__ csq, float* __restrict__ out) {

  // LDS: 16KB + 16KB + 10KB + 10KB = 52KB
  __shared__ v16h cenH[2 * 8 * 32];            // centers chunk, f16 hi, B-frag order
  __shared__ v16h cenL[2 * 8 * 32];            // centers chunk, f16 lo
  __shared__ _Float16 Wl[OUTDIM * 40];         // W chunk [o][m], stride 40 (pad)
  __shared__ _Float16 phib[WAVES * 16 * 40];   // per-wave phi transpose buffer

  const int tid  = threadIdx.x;
  const int lane = tid & 31;
  const int wv   = tid >> 5;
  const int r16  = lane & 15;
  const int hl   = lane >> 4;
  const int n0   = blockIdx.x * ROWS_PER_WG + wv * 16;

  // ---- x A-fragments (16 rows x 256 K), f16 hi/lo, resident in VGPRs ----
  // A layout (16-bit, 16x32): lane<16 holds row=lane, K in {8h..8h+7} u {16+8h..}
  v16h xh[8], xl[8];
  {
    const float* xr = x + (size_t)(n0 + r16) * INDIM;
    #pragma unroll
    for (int s = 0; s < 8; ++s) {
      const float* p0 = xr + 32 * s + 8 * hl;
      float4 a0 = *(const float4*)(p0);
      float4 a1 = *(const float4*)(p0 + 4);
      float4 b0 = *(const float4*)(p0 + 16);
      float4 b1 = *(const float4*)(p0 + 20);
      float f[16] = {a0.x,a0.y,a0.z,a0.w, a1.x,a1.y,a1.z,a1.w,
                     b0.x,b0.y,b0.z,b0.w, b1.x,b1.y,b1.z,b1.w};
      v16h hi, lo;
      #pragma unroll
      for (int e = 0; e < 16; ++e) {
        _Float16 h = (_Float16)f[e];
        hi[e] = h;
        lo[e] = (_Float16)(f[e] - (float)h);
      }
      xh[s] = hi; xl[s] = lo;
    }
  }

  // per-lane ||x||^2 for the 8 phi rows this lane sees (C-layout rows v+8*hl)
  float xsv[8];
  #pragma unroll
  for (int v = 0; v < 8; ++v) xsv[v] = xsq[n0 + v + 8 * hl];

  // accumulators pre-initialized with bias
  v8f acc[8];
  #pragma unroll
  for (int t = 0; t < 8; ++t) {
    float bv = bias[t * 16 + r16];
    v8f a;
    #pragma unroll
    for (int v = 0; v < 8; ++v) a[v] = bv;
    acc[t] = a;
  }

  _Float16* phw = phib + wv * (16 * 40);

  for (int ch = 0; ch < NCHUNK; ++ch) {
    const int m0 = ch * MT;
    __syncthreads();   // previous chunk fully consumed before restage

    // ---- stage centers chunk: 32 rows x 256 K, f32 -> f16 hi/lo, swizzled.
    // B layout (16-bit, 32x16): lane<16 = col, holds K 0..7,16..23; lane>=16 K 8..15,24..31.
    #pragma unroll
    for (int it = 0; it < 4; ++it) {
      int flat = it * WG_THREADS + tid;        // 0..1023
      int row  = flat >> 5;                    // 0..31 (center within chunk)
      int kr   = (flat & 31) * 8;              // 0..248, coalesced along K
      const float* src = centers + (size_t)(m0 + row) * INDIM + kr;
      float4 f0 = *(const float4*)src;
      float4 f1 = *(const float4*)(src + 4);
      float f[8] = {f0.x,f0.y,f0.z,f0.w, f1.x,f1.y,f1.z,f1.w};
      H8U Hu, Lu;
      #pragma unroll
      for (int e = 0; e < 8; ++e) {
        _Float16 h = (_Float16)f[e];
        Hu.h[e] = h;
        Lu.h[e] = (_Float16)(f[e] - (float)h);
      }
      int T = row >> 4, lr = row & 15;
      int s = kr >> 5, hh = (kr >> 3) & 1, e0 = ((kr >> 4) & 1) * 8;
      int fi = ((T * 8 + s) * 32 + (hh * 16 + lr)) * 16 + e0;  // f16 index
      *(uint4*)((_Float16*)cenH + fi) = Hu.u;
      *(uint4*)((_Float16*)cenL + fi) = Lu.u;
    }

    // ---- stage W chunk: Wl[o][m] = W[o][m0+m], f16, stride 40 ----
    {
      int o  = tid >> 1;
      int mh = (tid & 1) * 16;
      const float* ws = W + (size_t)o * NCENT + m0 + mh;
      float4 w0 = *(const float4*)(ws);
      float4 w1 = *(const float4*)(ws + 4);
      float4 w2 = *(const float4*)(ws + 8);
      float4 w3 = *(const float4*)(ws + 12);
      float g[16] = {w0.x,w0.y,w0.z,w0.w, w1.x,w1.y,w1.z,w1.w,
                     w2.x,w2.y,w2.z,w2.w, w3.x,w3.y,w3.z,w3.w};
      H8U A, B;
      #pragma unroll
      for (int e = 0; e < 8; ++e) { A.h[e] = (_Float16)g[e]; B.h[e] = (_Float16)g[8 + e]; }
      *(uint4*)(Wl + o * 40 + mh)     = A.u;
      *(uint4*)(Wl + o * 40 + mh + 8) = B.u;
    }
    __syncthreads();

    // ---- cross = x @ c^T via f16x3 WMMA; then phi = exp2(-sig*lg2e*sqdist)
    #pragma unroll
    for (int T = 0; T < 2; ++T) {
      v8f cr = {};
      #pragma unroll
      for (int s = 0; s < 8; ++s) {
        v16h bh = cenH[(T * 8 + s) * 32 + lane];
        v16h bl = cenL[(T * 8 + s) * 32 + lane];
        cr = __builtin_amdgcn_wmma_f32_16x16x32_f16(false, xh[s], false, bh, (short)0, cr, false, false);
        cr = __builtin_amdgcn_wmma_f32_16x16x32_f16(false, xh[s], false, bl, (short)0, cr, false, false);
        cr = __builtin_amdgcn_wmma_f32_16x16x32_f16(false, xl[s], false, bh, (short)0, cr, false, false);
      }
      int mg = m0 + T * 16 + r16;
      float cs = csq[mg];
      float sg = sigmas[mg] * 1.4426950408889634f;  // sigma * log2(e)
      #pragma unroll
      for (int v = 0; v < 8; ++v) {
        float sq = fmaf(-2.0f, cr[v], xsv[v] + cs);
        float ph = exp2f(-sg * sq);
        // C-layout (row v+8*hl, col r16) -> row-major f16 scratch (A-major)
        phw[(v + 8 * hl) * 40 + T * 16 + r16] = (_Float16)ph;
      }
    }

    // ---- GEMM2: acc[t] += phi(16x32) @ W^T(32x16), f16 WMMA ----
    // same-wave LDS store->load is in-order; no workgroup barrier needed
    V16U pa;
    pa.u[0] = *(const uint4*)(phw + r16 * 40 + 16 * hl);
    pa.u[1] = *(const uint4*)(phw + r16 * 40 + 16 * hl + 8);
    #pragma unroll
    for (int t = 0; t < 8; ++t) {
      const _Float16* wp = Wl + (t * 16 + r16) * 40 + 16 * hl;
      V16U wb;
      wb.u[0] = *(const uint4*)(wp);
      wb.u[1] = *(const uint4*)(wp + 8);
      acc[t] = __builtin_amdgcn_wmma_f32_16x16x32_f16(false, pa.v, false, wb.v, (short)0, acc[t], false, false);
    }
  }

  // ---- store out[n, o] from C-layout accumulators ----
  #pragma unroll
  for (int t = 0; t < 8; ++t) {
    #pragma unroll
    for (int v = 0; v < 8; ++v) {
      out[(size_t)(n0 + v + 8 * hl) * OUTDIM + t * 16 + r16] = acc[t][v];
    }
  }
}

extern "C" void kernel_launch(void* const* d_in, const int* in_sizes, int n_in,
                              void* d_out, int out_size, void* d_ws, size_t ws_size,
                              hipStream_t stream) {
  const float* x       = (const float*)d_in[0];
  const float* centers = (const float*)d_in[1];
  const float* sigmas  = (const float*)d_in[2];
  const float* W       = (const float*)d_in[3];
  const float* b       = (const float*)d_in[4];
  float* out = (float*)d_out;

  float* xsq = (float*)d_ws;            // 65536 floats
  float* csq = xsq + N_ROWS;            // 1024 floats  (needs 266,240 B of ws)

  rbf_sq_kernel<<<(N_ROWS + NCENT) / 8, WG_THREADS, 0, stream>>>(x, centers, xsq, csq);
  rbf_fused_kernel<<<N_ROWS / ROWS_PER_WG, WG_THREADS, 0, stream>>>(
      x, centers, sigmas, W, b, xsq, csq, out);
}
